// ResCLIPAttention_41781441856257
// MI455X (gfx1250) — compile-verified, matrix operand order checked by reading, not run on hardware
//
#include <hip/hip_runtime.h>
#include <cstdint>

#define NHEADS 16
#define HD     64
#define BATCH  4
#define SEQ    1032
#define CDIM   1024
#define NPAD   1088          // SEQ padded to multiple of 64 (two-stage PV k-loop)
#define NTILES 65            // ceil(SEQ/16)
#define SCALE  0.125f        // hd^-0.5
#define TAU    2.0f
#define LAM    0.6f

typedef __attribute__((ext_vector_type(16))) __bf16 v16bf;
typedef __attribute__((ext_vector_type(8)))  float  v8f;

__device__ __forceinline__ uint16_t f2bf(float f) {
  uint32_t u = __float_as_uint(f);
  uint32_t r = u + 0x7FFFu + ((u >> 16) & 1u);   // round-to-nearest-even
  return (uint16_t)(r >> 16);
}

// Load one 16x32 bf16 WMMA fragment (A-layout / mirrored B-layout, ISA 7.12.2):
// lane holds row (lane&15); VGPR v holds the bf16 pair at
//   k = (v>=4 ? 16 : 0) + (lane>=16 ? 8 : 0) + (v&3)*2
// rowp must point at [row][k0] of a row-major bf16 matrix (4B aligned).
// Per lane this is two contiguous 16B chunks -> compiler emits 2x load_b128.
__device__ __forceinline__ v16bf frag_ld(const uint16_t* __restrict__ rowp, int khalf) {
  uint32_t u[8];
#pragma unroll
  for (int v = 0; v < 8; ++v) {
    int k = ((v & 4) << 2) + khalf * 8 + (v & 3) * 2;
    u[v] = *reinterpret_cast<const uint32_t*>(rowp + k);
  }
  v16bf r;
  __builtin_memcpy(&r, u, sizeof(r));
  return r;
}

__device__ __forceinline__ v8f bwmma(v16bf a, v16bf b, v8f c) {
  return __builtin_amdgcn_wmma_f32_16x16x32_bf16(false, a, false, b, (short)0, c,
                                                 false, false);
}

// ------------------------------------------------- fp32 -> bf16 (4 elems/thread)
__global__ void cvt_bf16x4(const float* __restrict__ src, uint16_t* __restrict__ dst,
                           int n4) {
  int i = blockIdx.x * blockDim.x + threadIdx.x;
  if (i < n4) {
    const float4 f = reinterpret_cast<const float4*>(src)[i];
    union { uint16_t h[4]; uint64_t q; } u;
    u.h[0] = f2bf(f.x); u.h[1] = f2bf(f.y); u.h[2] = f2bf(f.z); u.h[3] = f2bf(f.w);
    reinterpret_cast<uint64_t*>(dst)[i] = u.q;
  }
}

// --------------------------------------------- K/V GEMM: x @ Wkv^T (+bias)
// Writes K in head layout [bh][n][hd] and V transposed [bh][hd][NPAD] (bf16).
__global__ void __launch_bounds__(128)
kv_gemm(const uint16_t* __restrict__ xb, const uint16_t* __restrict__ wkv,
        const float* __restrict__ qkv_b, uint16_t* __restrict__ Kh,
        uint16_t* __restrict__ Vt) {
  const int lane  = threadIdx.x & 31;
  const int wv    = threadIdx.x >> 5;
  const int khalf = lane >> 4;
  const int nl    = lane & 15;
  const int g     = blockIdx.x * 4 + wv;      // 64-channel group in [0,2048)
  const int n0    = blockIdx.y * 16;
  const int b     = blockIdx.z;
  const int d0    = g * 64;

  const int arow = min(n0 + nl, SEQ - 1);
  const uint16_t* xrow = xb + ((size_t)b * SEQ + arow) * CDIM;
  const uint16_t* wr[4];
#pragma unroll
  for (int j = 0; j < 4; ++j)
    wr[j] = wkv + (size_t)(d0 + j * 16 + nl) * CDIM;

  // Two-stage ping-pong pipeline (no register rotation copies).
  v8f acc[4] = {};
  v16bf a0 = frag_ld(xrow, khalf);
  v16bf b0[4];
#pragma unroll
  for (int j = 0; j < 4; ++j) b0[j] = frag_ld(wr[j], khalf);

  for (int kk = 0; kk < CDIM; kk += 64) {
    v16bf a1 = frag_ld(xrow + kk + 32, khalf);          // stage-1 loads
    v16bf b1[4];
#pragma unroll
    for (int j = 0; j < 4; ++j) b1[j] = frag_ld(wr[j] + kk + 32, khalf);
#pragma unroll
    for (int j = 0; j < 4; ++j) acc[j] = bwmma(a0, b0[j], acc[j]);  // stage-0 math

    const int kn = (kk + 64 < CDIM) ? kk + 64 : 0;      // harmless wrap on last iter
    a0 = frag_ld(xrow + kn, khalf);                     // next stage-0 loads
#pragma unroll
    for (int j = 0; j < 4; ++j) b0[j] = frag_ld(wr[j] + kn, khalf);
#pragma unroll
    for (int j = 0; j < 4; ++j) acc[j] = bwmma(a1, b1[j], acc[j]);  // stage-1 math
  }

#pragma unroll
  for (int v = 0; v < 8; ++v) {
    int row = n0 + khalf * 8 + v;
    if (row >= SEQ) continue;
#pragma unroll
    for (int j = 0; j < 4; ++j) {
      int d = d0 + j * 16 + nl;                 // channel in [0,2048)
      uint16_t bv = f2bf(acc[j][v] + qkv_b[CDIM + d]);
      if (d < CDIM) {                           // K part
        int h = d >> 6, e = d & 63;
        Kh[(((size_t)(b * NHEADS + h)) * SEQ + row) * HD + e] = bv;
      } else {                                  // V part (transposed store)
        int dv = d - CDIM, h = dv >> 6, e = dv & 63;
        Vt[(((size_t)(b * NHEADS + h)) * HD + e) * NPAD + row] = bv;
      }
    }
  }
}

// ---------------- Fused attention: S = K Kt, softmax, rcs blend, O = P V
__global__ void __launch_bounds__(128)
attn_fused(const uint16_t* __restrict__ Kh, const uint16_t* __restrict__ Vt,
           const float* __restrict__ addm, const float* __restrict__ rcs,
           uint16_t* __restrict__ AO) {
  extern __shared__ char smem[];
  float*    Sm = (float*)smem;                    // 16 x NPAD f32
  uint16_t* Pm = (uint16_t*)(Sm + 16 * NPAD);     // 16 x NPAD bf16

  const int tid   = threadIdx.x;
  const int lane  = tid & 31;
  const int wv    = tid >> 5;
  const int khalf = lane >> 4;
  const int nl    = lane & 15;
  const int i0    = blockIdx.x * 16;
  const int bh    = blockIdx.y;
  const uint16_t* Kb = Kh + (size_t)bh * SEQ * HD;

  // Hint the streamed rcs rows into cache while we do KK^T.
  {
    int r = tid >> 3, sub = tid & 7, row = i0 + r;
    if (row < SEQ) {
      const float* rr = rcs + ((size_t)bh * SEQ + row) * SEQ;
      for (int c = sub * 132; c < SEQ; c += 1056) __builtin_prefetch(rr + c, 0, 1);
    }
  }

  // Phase 1: S(16 x SEQ) = K_tile(16x64) . K^T   (wave-parallel over col tiles)
  // A fragments are loop-invariant: load the 16 query rows once.
  const int ar = min(i0 + nl, SEQ - 1);
  const uint16_t* arow = Kb + (size_t)ar * HD;
  const v16bf a0 = frag_ld(arow,      khalf);
  const v16bf a1 = frag_ld(arow + 32, khalf);
  for (int j = wv; j < NTILES; j += 4) {
    int br = min(j * 16 + nl, SEQ - 1);
    const uint16_t* brow = Kb + (size_t)br * HD;
    v16bf b0 = frag_ld(brow,      khalf);
    v16bf b1 = frag_ld(brow + 32, khalf);
    v8f acc = {};
    acc = bwmma(a0, b0, acc);
    acc = bwmma(a1, b1, acc);
#pragma unroll
    for (int v = 0; v < 8; ++v)
      Sm[(khalf * 8 + v) * NPAD + j * 16 + nl] = acc[v];
  }
  __syncthreads();

  // Phase 2: row softmax (8 lanes per row), then blend with rcs -> bf16 P.
  {
    int r = tid >> 3, sub = tid & 7, row = i0 + r;
    float*    srow = Sm + r * NPAD;
    uint16_t* prow = Pm + r * NPAD;
    if (row < SEQ) {
      const float* adr = addm + (size_t)row * SEQ;
      const float* rr  = rcs + ((size_t)bh * SEQ + row) * SEQ;
      float mx = -3.4e38f;
      for (int c = sub; c < SEQ; c += 8) {
        float t = fmaf(srow[c], SCALE, adr[c]);
        srow[c] = t;
        mx = fmaxf(mx, t);
      }
      mx = fmaxf(mx, __shfl_xor(mx, 1, 32));
      mx = fmaxf(mx, __shfl_xor(mx, 2, 32));
      mx = fmaxf(mx, __shfl_xor(mx, 4, 32));
      float sm = 0.f;
      for (int c = sub; c < SEQ; c += 8) {
        float e = __expf(srow[c] - mx);
        srow[c] = e;
        sm += e;
      }
      sm += __shfl_xor(sm, 1, 32);
      sm += __shfl_xor(sm, 2, 32);
      sm += __shfl_xor(sm, 4, 32);
      float cs = (TAU * (1.f - LAM)) / sm;
      for (int c = sub; c < NPAD; c += 8)
        prow[c] = (c < SEQ) ? f2bf(fmaf(cs, srow[c], (TAU * LAM) * rr[c]))
                            : (uint16_t)0;
    } else {
      for (int c = sub; c < NPAD; c += 8) prow[c] = 0;  // dead rows -> P = 0
    }
  }
  __syncthreads();

  // Phase 3: O(16x64) = P(16xNPAD) . V ; wave wv owns hd columns [wv*16, wv*16+16)
  {
    const uint16_t* vrow = Vt + ((size_t)bh * HD + wv * 16 + nl) * NPAD;
    const uint16_t* pr   = Pm + (size_t)nl * NPAD;
    v8f acc = {};
    v16bf pa0 = frag_ld(pr,   khalf);
    v16bf vb0 = frag_ld(vrow, khalf);
    for (int kk = 0; kk < NPAD; kk += 64) {             // NPAD % 64 == 0
      v16bf pa1 = frag_ld(pr + kk + 32,   khalf);
      v16bf vb1 = frag_ld(vrow + kk + 32, khalf);
      acc = bwmma(pa0, vb0, acc);
      const int kn = (kk + 64 < NPAD) ? kk + 64 : 0;
      pa0 = frag_ld(pr + kn,   khalf);
      vb0 = frag_ld(vrow + kn, khalf);
      acc = bwmma(pa1, vb1, acc);
    }
    const int b = bh >> 4, h = bh & 15;
#pragma unroll
    for (int v = 0; v < 8; ++v) {
      int row = i0 + khalf * 8 + v;
      if (row < SEQ)
        AO[((size_t)b * SEQ + row) * CDIM + h * 64 + wv * 16 + nl] = f2bf(acc[v]);
    }
  }
}

// ----------------------------------- Projection: out = AO @ proj_w^T + proj_b
__global__ void __launch_bounds__(128)
proj_gemm(const uint16_t* __restrict__ ao, const uint16_t* __restrict__ pw,
          const float* __restrict__ pb, float* __restrict__ out) {
  const int lane  = threadIdx.x & 31;
  const int wv    = threadIdx.x >> 5;
  const int khalf = lane >> 4;
  const int nl    = lane & 15;
  const int g     = blockIdx.x * 4 + wv;     // 64-col group in [0,1024)
  const int r0    = blockIdx.y * 16;         // B*SEQ = 4128 = 258*16 exactly
  const int d0    = g * 64;

  const uint16_t* arow = ao + (size_t)(r0 + nl) * CDIM;
  const uint16_t* wr[4];
#pragma unroll
  for (int j = 0; j < 4; ++j)
    wr[j] = pw + (size_t)(d0 + j * 16 + nl) * CDIM;

  v8f acc[4] = {};
  v16bf a0 = frag_ld(arow, khalf);
  v16bf b0[4];
#pragma unroll
  for (int j = 0; j < 4; ++j) b0[j] = frag_ld(wr[j], khalf);

  for (int kk = 0; kk < CDIM; kk += 64) {
    v16bf a1 = frag_ld(arow + kk + 32, khalf);
    v16bf b1[4];
#pragma unroll
    for (int j = 0; j < 4; ++j) b1[j] = frag_ld(wr[j] + kk + 32, khalf);
#pragma unroll
    for (int j = 0; j < 4; ++j) acc[j] = bwmma(a0, b0[j], acc[j]);

    const int kn = (kk + 64 < CDIM) ? kk + 64 : 0;
    a0 = frag_ld(arow + kn, khalf);
#pragma unroll
    for (int j = 0; j < 4; ++j) b0[j] = frag_ld(wr[j] + kn, khalf);
#pragma unroll
    for (int j = 0; j < 4; ++j) acc[j] = bwmma(a1, b1[j], acc[j]);
  }

#pragma unroll
  for (int v = 0; v < 8; ++v) {
    int row = r0 + khalf * 8 + v;
#pragma unroll
    for (int j = 0; j < 4; ++j) {
      int d = d0 + j * 16 + nl;
      out[(size_t)row * CDIM + d] = acc[j][v] + pb[d];
    }
  }
}

// ---------------------------------------------------------------- launcher
extern "C" void kernel_launch(void* const* d_in, const int* in_sizes, int n_in,
                              void* d_out, int out_size, void* d_ws, size_t ws_size,
                              hipStream_t stream) {
  const float* x      = (const float*)d_in[0];
  const float* qkv_w  = (const float*)d_in[1];
  const float* qkv_b  = (const float*)d_in[2];
  const float* proj_w = (const float*)d_in[3];
  const float* proj_b = (const float*)d_in[4];
  const float* addm   = (const float*)d_in[5];
  const float* rcs    = (const float*)d_in[6];
  float* out = (float*)d_out;

  char* ws = (char*)d_ws;
  const size_t N_X  = (size_t)BATCH * SEQ * CDIM;        // 4,227,072
  const size_t N_W  = (size_t)2 * CDIM * CDIM;           // 2,097,152
  const size_t N_P  = (size_t)CDIM * CDIM;               // 1,048,576
  uint16_t* XB  = (uint16_t*)(ws);                                   // x  bf16
  uint16_t* WKV = (uint16_t*)(ws + N_X * 2);                         // Wk|Wv bf16
  uint16_t* PW  = (uint16_t*)(ws + N_X * 2 + N_W * 2);               // proj_w bf16
  uint16_t* KH  = (uint16_t*)(ws + N_X * 2 + N_W * 2 + N_P * 2);     // [bh][n][hd]
  uint16_t* VT  = KH + (size_t)BATCH * NHEADS * SEQ * HD;            // [bh][hd][NPAD]
  uint16_t* AO  = VT + (size_t)BATCH * NHEADS * HD * NPAD;           // [b][n][C] bf16

  cvt_bf16x4<<<(int)((N_X / 4 + 255) / 256), 256, 0, stream>>>(x, XB, (int)(N_X / 4));
  cvt_bf16x4<<<(int)((N_W / 4 + 255) / 256), 256, 0, stream>>>(
      qkv_w + (size_t)CDIM * CDIM, WKV, (int)(N_W / 4));
  cvt_bf16x4<<<(int)((N_P / 4 + 255) / 256), 256, 0, stream>>>(proj_w, PW,
                                                               (int)(N_P / 4));

  kv_gemm<<<dim3(8, NTILES, BATCH), 128, 0, stream>>>(XB, WKV, qkv_b, KH, VT);

  const size_t smem = (size_t)16 * NPAD * 4 + (size_t)16 * NPAD * 2;  // ~102 KB
  hipFuncSetAttribute(reinterpret_cast<const void*>(attn_fused),
                      hipFuncAttributeMaxDynamicSharedMemorySize, (int)smem);
  attn_fused<<<dim3(NTILES, BATCH * NHEADS), 128, smem, stream>>>(KH, VT, addm,
                                                                  rcs, AO);

  proj_gemm<<<dim3(4, 258), 128, 0, stream>>>(AO, PW, proj_b, out);
}
